// RobustGCN_18047452578190
// MI455X (gfx1250) — matrix operand order repaired and checked by reading, not compiled, not run on hardware
//
#include <hip/hip_runtime.h>
#include <hip/hip_bf16.h>
#include <math.h>

// ---------------------------------------------------------------------------
// RobustGCN forward for MI455X (gfx1250, wave32, WMMA).
//   d_in: x[0], edge_index[1](int64 2xE), noise[2], Wm0[3], bm0[4], Wv0[5],
//         bv0[6], Wm1[7], bm1[8], Wv1[9], bv1[10]
//   d_out: N*40 f32 (log_softmax)
// ---------------------------------------------------------------------------

#define NNODES 100000
#define NEDGES 1600000
#define FIN    128
#define FHID   64
#define FOUT   40
#define FOUTP  48          // 40 padded to 3 WMMA n-tiles

typedef __attribute__((ext_vector_type(16))) _Float16 v16h;
typedef __attribute__((ext_vector_type(8)))  _Float16 v8h;
typedef __attribute__((ext_vector_type(8)))  float    v8f;

// ---- fragment loaders (ISA 7.12.2 layouts) --------------------------------
// A (16x32 f16, MxK): lane = grp*16+m ; halves e=0..7 -> K=k0+grp*8+e,
//                     e=8..15 -> K=k0+16+grp*8+(e-8). Two contiguous 16B loads.
__device__ __forceinline__ v16h load_frag_a(const _Float16* p) {
  union { v16h v; v8h h[2]; } u;
  u.h[0] = *(const v8h*)(p);
  u.h[1] = *(const v8h*)(p + 16);
  return u.v;
}
// B (32x16 f16, KxN) from transposed (N-major) weights: lane = grp*16+n ;
// halves e=0..7 -> K=k0+grp*8+e, e=8..15 -> K=k0+16+grp*8+(e-8). Two 16B loads.
__device__ __forceinline__ v16h load_frag_b(const _Float16* p) {
  union { v16h v; v8h h[2]; } u;
  u.h[0] = *(const v8h*)(p);
  u.h[1] = *(const v8h*)(p + 16);
  return u.v;
}

__device__ __forceinline__ float elu_f(float x) { return x > 0.f ? x : expm1f(x); }

// ---- setup kernels --------------------------------------------------------
__global__ void k_convert_x(const float* __restrict__ x, _Float16* __restrict__ xh) {
  int i = blockIdx.x * blockDim.x + threadIdx.x;
  if (i < NNODES * FIN) xh[i] = (_Float16)x[i];
}

// Transpose+convert weights to out-major f16; pad layer-1 out dim to 48.
__global__ void k_convert_w(const float* __restrict__ Wm0, const float* __restrict__ Wv0,
                            const float* __restrict__ Wm1, const float* __restrict__ Wv1,
                            _Float16* __restrict__ Wm0t, _Float16* __restrict__ Wv0t,
                            _Float16* __restrict__ Wm1t, _Float16* __restrict__ Wv1t) {
  int i = blockIdx.x * blockDim.x + threadIdx.x;
  if (i < FIN * FHID) {                       // layer 0: [128 x 64] -> t[64][128]
    int k = i / FHID, n = i % FHID;
    Wm0t[(size_t)n * FIN + k] = (_Float16)Wm0[i];
    Wv0t[(size_t)n * FIN + k] = (_Float16)Wv0[i];
  }
  if (i < FHID * FOUTP) {                     // layer 1: [64 x 40]->t[48][64], pad 0
    int k = i / FOUTP, n = i % FOUTP;
    float m = (n < FOUT) ? Wm1[k * FOUT + n] : 0.f;
    float v = (n < FOUT) ? Wv1[k * FOUT + n] : 0.f;
    Wm1t[(size_t)n * FHID + k] = (_Float16)m;
    Wv1t[(size_t)n * FHID + k] = (_Float16)v;
  }
}

__global__ void k_deg_init(float* __restrict__ deg) {
  int i = blockIdx.x * blockDim.x + threadIdx.x;
  if (i < NNODES) deg[i] = 1.0f;              // self loop
}

__global__ void k_deg_count(const long long* __restrict__ ei, float* __restrict__ deg) {
  int e = blockIdx.x * blockDim.x + threadIdx.x;
  if (e < NEDGES) unsafeAtomicAdd(&deg[(int)ei[e]], 1.0f);
}

__global__ void k_dis(const float* __restrict__ deg, float* __restrict__ dis0,
                      float* __restrict__ dis1) {
  int i = blockIdx.x * blockDim.x + threadIdx.x;
  if (i < NNODES) {
    float d = deg[i];
    dis0[i] = d > 0.f ? rsqrtf(d) : 0.f;      // d^-1/2
    dis1[i] = d > 0.f ? 1.0f / d : 0.f;       // d^-1
  }
}

// ---- layer 0: hm = elu(x@Wm0+b), hv = relu(x@Wv0+b)  [WMMA f16->f32] ------
__global__ void k_gemm0(const _Float16* __restrict__ xh,
                        const _Float16* __restrict__ Wm0t, const _Float16* __restrict__ Wv0t,
                        const float* __restrict__ bm0, const float* __restrict__ bv0,
                        _Float16* __restrict__ hm, _Float16* __restrict__ hv) {
  int wid  = (blockIdx.x * blockDim.x + threadIdx.x) >> 5;
  int lane = threadIdx.x & 31;
  int mtile = wid >> 2;                       // 6250 m-tiles
  int ntile = wid & 3;                        // 4 n-tiles (64 cols)
  if (mtile >= NNODES / 16) return;
  int m0 = mtile * 16, n0 = ntile * 16;
  int grp = lane >> 4, mr = lane & 15;

  v8f cm = {}; v8f cv = {};
  const _Float16* ap  = xh   + (size_t)(m0 + mr) * FIN + grp * 8;
  const _Float16* bmp = Wm0t + (size_t)(n0 + mr) * FIN + grp * 8;
  const _Float16* bvp = Wv0t + (size_t)(n0 + mr) * FIN + grp * 8;
#pragma unroll
  for (int k0 = 0; k0 < FIN; k0 += 32) {
    v16h a  = load_frag_a(ap + k0);
    v16h bm = load_frag_b(bmp + k0);
    v16h bv = load_frag_b(bvp + k0);
    cm = __builtin_amdgcn_wmma_f32_16x16x32_f16(false, a, false, bm, (short)0, cm, false, false);
    cv = __builtin_amdgcn_wmma_f32_16x16x32_f16(false, a, false, bv, (short)0, cv, false, false);
  }
  int n = n0 + mr;
  float bbm = bm0[n], bbv = bv0[n];
#pragma unroll
  for (int r = 0; r < 8; ++r) {               // D: row = m0 + r + 8*grp, col = n
    int row = m0 + r + grp * 8;
    float me = elu_f(cm[r] + bbm);
    float va = fmaxf(cv[r] + bbv, 0.f);
    hm[(size_t)row * FHID + n] = (_Float16)me;
    hv[(size_t)row * FHID + n] = (_Float16)va;
  }
}

// ---- layer 1 + attenuation epilogue ---------------------------------------
__global__ void k_gemm1(const _Float16* __restrict__ hm, const _Float16* __restrict__ hv,
                        const _Float16* __restrict__ Wm1t, const _Float16* __restrict__ Wv1t,
                        const float* __restrict__ bm1, const float* __restrict__ bv1,
                        float* __restrict__ m1, float* __restrict__ v1) {
  int wid  = (blockIdx.x * blockDim.x + threadIdx.x) >> 5;
  int lane = threadIdx.x & 31;
  int mtile = wid / 3;                        // 6250 m-tiles
  int ntile = wid % 3;                        // 3 n-tiles (48 padded cols)
  if (mtile >= NNODES / 16) return;
  int m0 = mtile * 16, n0 = ntile * 16;
  int grp = lane >> 4, mr = lane & 15;

  v8f cm = {}; v8f cv = {};
  const _Float16* amp = hm   + (size_t)(m0 + mr) * FHID + grp * 8;
  const _Float16* avp = hv   + (size_t)(m0 + mr) * FHID + grp * 8;
  const _Float16* bmp = Wm1t + (size_t)(n0 + mr) * FHID + grp * 8;
  const _Float16* bvp = Wv1t + (size_t)(n0 + mr) * FHID + grp * 8;
#pragma unroll
  for (int k0 = 0; k0 < FHID; k0 += 32) {
    v16h am = load_frag_a(amp + k0);
    v16h av = load_frag_a(avp + k0);
    v16h bm = load_frag_b(bmp + k0);
    v16h bv = load_frag_b(bvp + k0);
    cm = __builtin_amdgcn_wmma_f32_16x16x32_f16(false, am, false, bm, (short)0, cm, false, false);
    cv = __builtin_amdgcn_wmma_f32_16x16x32_f16(false, av, false, bv, (short)0, cv, false, false);
  }
  int n = n0 + mr;
  float bbm = (n < FOUT) ? bm1[n] : 0.f;
  float bbv = (n < FOUT) ? bv1[n] : 0.f;
#pragma unroll
  for (int r = 0; r < 8; ++r) {
    int row = m0 + r + grp * 8;
    float me  = elu_f(cm[r] + bbm);
    float va  = fmaxf(cv[r] + bbv, 0.f) + 1e-6f;
    float att = __expf(-va);
    m1[(size_t)row * FOUTP + n] = me * att;
    v1[(size_t)row * FOUTP + n] = va * att * att;
  }
}

// ---- SpMM scatter: agg[row] += w * feat[col]  (L2-resident fp32 atomics) --
__global__ void k_spmm(const long long* __restrict__ ei,
                       const float* __restrict__ dis0, const float* __restrict__ dis1,
                       const float* __restrict__ m1, const float* __restrict__ v1,
                       float* __restrict__ aggm, float* __restrict__ aggv) {
  long long gid = (long long)blockIdx.x * blockDim.x + threadIdx.x;
  const long long tot = (long long)(NEDGES + NNODES) * 8;
  if (gid >= tot) return;
  int part = (int)(gid & 7);                  // 8 threads/edge, 5 cols each
  long long e = gid >> 3;
  int r, c;
  if (e < NEDGES) { r = (int)ei[e]; c = (int)ei[NEDGES + e]; }
  else            { r = c = (int)(e - NEDGES); }  // self loop
  float w0 = dis0[r] * dis0[c];
  float w1 = dis1[r] * dis1[c];
  const float* ms = m1 + (size_t)c * FOUTP;
  const float* vs = v1 + (size_t)c * FOUTP;
  float* md = aggm + (size_t)r * FOUT;
  float* vd = aggv + (size_t)r * FOUT;
#pragma unroll
  for (int j = 0; j < 5; ++j) {
    int col = part * 5 + j;
    unsafeAtomicAdd(md + col, w0 * ms[col]);
    unsafeAtomicAdd(vd + col, w1 * vs[col]);
  }
}

// ---- out = log_softmax(aggm + noise*sqrt(aggv)) ; one wave32 per row ------
__global__ void k_final(const float* __restrict__ aggm, const float* __restrict__ aggv,
                        const float* __restrict__ noise, float* __restrict__ out) {
  int wid  = (blockIdx.x * blockDim.x + threadIdx.x) >> 5;
  int lane = threadIdx.x & 31;
  if (wid >= NNODES) return;
  size_t base = (size_t)wid * FOUT;
  int c1 = lane + 32;
  float x0 = aggm[base + lane] + noise[base + lane] * sqrtf(fmaxf(aggv[base + lane], 0.f));
  float x1 = -INFINITY;
  if (c1 < FOUT)
    x1 = aggm[base + c1] + noise[base + c1] * sqrtf(fmaxf(aggv[base + c1], 0.f));
  float mx = fmaxf(x0, x1);
#pragma unroll
  for (int off = 16; off >= 1; off >>= 1) mx = fmaxf(mx, __shfl_xor(mx, off, 32));
  float s = __expf(x0 - mx) + ((c1 < FOUT) ? __expf(x1 - mx) : 0.f);
#pragma unroll
  for (int off = 16; off >= 1; off >>= 1) s += __shfl_xor(s, off, 32);
  float lse = mx + __logf(s);
  out[base + lane] = x0 - lse;
  if (c1 < FOUT) out[base + c1] = x1 - lse;
}

// ---------------------------------------------------------------------------
extern "C" void kernel_launch(void* const* d_in, const int* in_sizes, int n_in,
                              void* d_out, int out_size, void* d_ws, size_t ws_size,
                              hipStream_t stream) {
  const float*     x     = (const float*)d_in[0];
  const long long* ei    = (const long long*)d_in[1];
  const float*     noise = (const float*)d_in[2];
  const float*     Wm0   = (const float*)d_in[3];
  const float*     bm0   = (const float*)d_in[4];
  const float*     Wv0   = (const float*)d_in[5];
  const float*     bv0   = (const float*)d_in[6];
  const float*     Wm1   = (const float*)d_in[7];
  const float*     bm1   = (const float*)d_in[8];
  const float*     Wv1   = (const float*)d_in[9];
  const float*     bv1   = (const float*)d_in[10];
  float* out = (float*)d_out;

  // workspace carve-up (256B aligned)
  char* ws = (char*)d_ws;
  size_t off = 0;
  auto carve = [&](size_t bytes) -> char* {
    off = (off + 255) & ~(size_t)255;
    char* p = ws + off; off += bytes; return p;
  };
  _Float16* xh   = (_Float16*)carve((size_t)NNODES * FIN * 2);
  _Float16* Wm0t = (_Float16*)carve((size_t)FHID * FIN * 2);
  _Float16* Wv0t = (_Float16*)carve((size_t)FHID * FIN * 2);
  _Float16* Wm1t = (_Float16*)carve((size_t)FOUTP * FHID * 2);
  _Float16* Wv1t = (_Float16*)carve((size_t)FOUTP * FHID * 2);
  _Float16* hm   = (_Float16*)carve((size_t)NNODES * FHID * 2);
  _Float16* hv   = (_Float16*)carve((size_t)NNODES * FHID * 2);
  float*    m1   = (float*)carve((size_t)NNODES * FOUTP * 4);
  float*    v1   = (float*)carve((size_t)NNODES * FOUTP * 4);
  float*    aggm = (float*)carve((size_t)NNODES * FOUT * 4 * 2);  // aggm+aggv adjacent
  float*    aggv = aggm + (size_t)NNODES * FOUT;
  float*    deg  = (float*)carve((size_t)NNODES * 4);
  float*    dis0 = (float*)carve((size_t)NNODES * 4);
  float*    dis1 = (float*)carve((size_t)NNODES * 4);
  (void)ws_size; (void)in_sizes; (void)n_in; (void)out_size;

  const int T = 256;

  // setup / degree
  k_convert_x<<<(NNODES * FIN + T - 1) / T, T, 0, stream>>>(x, xh);
  k_convert_w<<<(FIN * FHID + T - 1) / T, T, 0, stream>>>(Wm0, Wv0, Wm1, Wv1,
                                                          Wm0t, Wv0t, Wm1t, Wv1t);
  k_deg_init<<<(NNODES + T - 1) / T, T, 0, stream>>>(deg);
  hipMemsetAsync(aggm, 0, (size_t)NNODES * FOUT * 4 * 2, stream);
  k_deg_count<<<(NEDGES + T - 1) / T, T, 0, stream>>>(ei, deg);
  k_dis<<<(NNODES + T - 1) / T, T, 0, stream>>>(deg, dis0, dis1);

  // WMMA GEMMs (wave = one 16x16 tile; 8 waves/block)
  int waves0 = (NNODES / 16) * 4;             // 25000
  k_gemm0<<<(waves0 + 7) / 8, T, 0, stream>>>(xh, Wm0t, Wv0t, bm0, bv0, hm, hv);
  int waves1 = (NNODES / 16) * 3;             // 18750
  k_gemm1<<<(waves1 + 7) / 8, T, 0, stream>>>(hm, hv, Wm1t, Wv1t, bm1, bv1, m1, v1);

  // edge scatter + final log-softmax
  long long sthreads = (long long)(NEDGES + NNODES) * 8;
  k_spmm<<<(int)((sthreads + T - 1) / T), T, 0, stream>>>(ei, dis0, dis1, m1, v1, aggm, aggv);
  k_final<<<(NNODES * 32 + T - 1) / T, T, 0, stream>>>(aggm, aggv, noise, out);
}
